// NaturalTransformer_81432579932597
// MI455X (gfx1250) — matrix operand order repaired and checked by reading, not compiled, not run on hardware
//
#include <hip/hip_runtime.h>
#include <hip/hip_bf16.h>

// ---------------- model dims ----------------
#define Bb 2
#define Ss 1024
#define Vv 32000
#define Ee 256
#define HIDh 512
#define Hh 8
#define Dd 64
#define Ll 6
#define MRP 128
#define Rr (2 * MRP + 1)   // 257
#define RP 320             // padded N (and row stride) for qrel, multiple of 64
#define Ff 2048
#define EPSf 1e-5f

typedef __attribute__((ext_vector_type(16))) __bf16 v16bf;
typedef __attribute__((ext_vector_type(8)))  __bf16 v8bf;
typedef __attribute__((ext_vector_type(2)))  __bf16 v2bf;
typedef __attribute__((ext_vector_type(8)))  float  v8f;
typedef __attribute__((ext_vector_type(2)))  float  v2f;   // trivial vector: OK through addrspace ptrs

// explicit global-address-space pointers: force global_load/global_store
// (pointer arrays otherwise defeat addrspace inference -> flat_load + dscnt waits)
#define GCF2 const __attribute__((address_space(1))) v2f*
#define GCF  const __attribute__((address_space(1))) float*
#define GF   __attribute__((address_space(1))) float*

// ---------------- embedding + sinusoidal positional encoding ----------------
__global__ __launch_bounds__(256)
void embed_pe_kernel(const int* __restrict__ x, const float* __restrict__ embed,
                     float* __restrict__ out)
{
    long long i = (long long)blockIdx.x * blockDim.x + threadIdx.x;
    if (i >= (long long)Bb * Ss * Ee) return;
    int e  = (int)(i % Ee);
    long long bs = i / Ee;
    int s  = (int)(bs % Ss);
    int tok = x[bs];
    int e2 = e & ~1;
    float div = __expf((float)e2 * (-9.210340371976184f / (float)Ee)); // -ln(10000)/E
    float ang = (float)s * div;
    float pe  = (e & 1) ? __cosf(ang) : __sinf(ang);
    out[i] = embed[(long long)tok * Ee + e] + pe;
}

// ---------------- zero-padded relT: relpad[l][r<257]=relT, else 0 ----------------
__global__ __launch_bounds__(256)
void relpad_kernel(const float* __restrict__ relE, float* __restrict__ relpad)
{
    int i = blockIdx.x * blockDim.x + threadIdx.x;
    if (i >= Ll * RP * Dd) return;
    int d = i % Dd;
    int r = (i / Dd) % RP;
    int l = i / (Dd * RP);
    relpad[i] = (r < Rr) ? relE[((long long)l * Rr + r) * Dd + d] : 0.f;
}

// ---------------- LayerNorm: one block per row ----------------
__global__ __launch_bounds__(256)
void layernorm_kernel(const float* __restrict__ in, const float* __restrict__ g,
                      const float* __restrict__ b, float* __restrict__ out, int N)
{
    __shared__ float red[256];
    long long row = blockIdx.x;
    const float* rp = in + row * N;
    float s = 0.f;
    for (int c = threadIdx.x; c < N; c += 256) s += rp[c];
    red[threadIdx.x] = s; __syncthreads();
    for (int st = 128; st > 0; st >>= 1) {
        if (threadIdx.x < st) red[threadIdx.x] += red[threadIdx.x + st];
        __syncthreads();
    }
    float mean = red[0] / (float)N; __syncthreads();
    float vs = 0.f;
    for (int c = threadIdx.x; c < N; c += 256) { float d = rp[c] - mean; vs += d * d; }
    red[threadIdx.x] = vs; __syncthreads();
    for (int st = 128; st > 0; st >>= 1) {
        if (threadIdx.x < st) red[threadIdx.x] += red[threadIdx.x + st];
        __syncthreads();
    }
    float inv = rsqrtf(red[0] / (float)N + EPSf);
    for (int c = threadIdx.x; c < N; c += 256)
        out[row * N + c] = (rp[c] - mean) * inv * g[c] + b[c];
}

// ---------------- fused relative-bias + softmax over one score row ----------------
__global__ __launch_bounds__(256)
void softmax_rel_kernel(float* __restrict__ scores, const float* __restrict__ qrel)
{
    __shared__ float red[256];
    int q = blockIdx.x;
    int z = blockIdx.y;
    int b = z / Hh, h = z % Hh;
    float* row = scores + ((long long)z * Ss + q) * Ss;
    const float* qr = qrel + (long long)(((long long)b * Ss + q) * Hh + h) * RP;
    const float sc = 0.125f; // 1/sqrt(D)
    float vals[4];
    float mx = -1e30f;
#pragma unroll
    for (int t = 0; t < 4; ++t) {
        int k = threadIdx.x + t * 256;
        int dd = k - q;
        dd = dd < -MRP ? -MRP : (dd > MRP ? MRP : dd);
        float v = (row[k] + qr[dd + MRP]) * sc;
        vals[t] = v;
        mx = fmaxf(mx, v);
    }
    red[threadIdx.x] = mx; __syncthreads();
    for (int st = 128; st > 0; st >>= 1) {
        if (threadIdx.x < st) red[threadIdx.x] = fmaxf(red[threadIdx.x], red[threadIdx.x + st]);
        __syncthreads();
    }
    mx = red[0]; __syncthreads();
    float sum = 0.f;
#pragma unroll
    for (int t = 0; t < 4; ++t) { vals[t] = __expf(vals[t] - mx); sum += vals[t]; }
    red[threadIdx.x] = sum; __syncthreads();
    for (int st = 128; st > 0; st >>= 1) {
        if (threadIdx.x < st) red[threadIdx.x] += red[threadIdx.x + st];
        __syncthreads();
    }
    float inv = 1.f / red[0];
#pragma unroll
    for (int t = 0; t < 4; ++t) row[threadIdx.x + t * 256] = vals[t] * inv;
}

// ---------------- batched/strided WMMA bf16 GEMM, 128x64x32 block tile ----------------
// C[z][m,n] = act( A[z][m,:] @ B[z][:,n] + bias[n] ) + resid[z][m,n]
// REQUIRES: M % 128 == 0, N % 64 == 0, K % 32 == 0, A k-contiguous (sAc == 1).
// BK1=true  : B k-contiguous (sBr == 1)   -> v2f pair loads along k
// BK1=false : B n-contiguous (sBc == 1)   -> lane-per-n coalesced loads
// Double-buffered LDS, one barrier per K-step. 8 wave32 waves in 4(M) x 2(N),
// each wave computes 32x32 output via 2x2 v_wmma_f32_16x16x32_bf16 tiles.
template<bool BK1>
__global__ __launch_bounds__(256)
void gemm_wmma_bf16(const float* __restrict__ A, const float* __restrict__ Bm,
                    const float* __restrict__ bias, const float* __restrict__ resid,
                    float* __restrict__ C,
                    int K, int Hdim,
                    long long sAb, long long sAh, long long sAr,
                    long long sBb, long long sBh, long long sBr, long long sBc,
                    long long sCb, long long sCh, long long sCr,
                    int act)
{
    __shared__ __bf16 As[2][128][40];   // [m][k], row stride 80B (16B-aligned runs)
    __shared__ __bf16 Bst[2][64][40];   // [n][k] transposed: fragment = contiguous run

    const int tid  = threadIdx.x;
    const int lane = tid & 31;
    const int wave = tid >> 5;
    const int lr   = lane & 15;
    const int hi   = lane >> 4;
    const int wm   = wave & 3;    // M sub-block (32 rows)
    const int wn   = wave >> 2;   // N sub-block (32 cols)

    const int m0 = blockIdx.y * 128;
    const int n0 = blockIdx.x * 64;
    const int z  = blockIdx.z;
    const int bz = z / Hdim, hz = z % Hdim;

    // ---- A staging: thread handles v2f pairs along k; 8 per K-step ----
    const int tg = tid >> 4;        // 0..15
    const int tp = (tid & 15) * 2;  // pair base col (k) 0..30
    const float* aBase = A + (long long)bz * sAb + (long long)hz * sAh;
    GCF2 paT[8];
#pragma unroll
    for (int t = 0; t < 8; ++t)
        paT[t] = (GCF2)(aBase + (long long)(m0 + tg + 16 * t) * sAr + tp);

    // ---- B staging pointers ----
    const float* bBase = Bm + (long long)bz * sBb + (long long)hz * sBh;
    GCF2 pbT[4];   // BK1 path (k-contiguous)
    GCF  pbS[8];   // !BK1 path (n-contiguous), pairs of k rows
    const int bc  = tid & 63;       // n index (coalesced across wave)
    const int bpr = tid >> 6;       // 0..3 k-pair group
#pragma unroll
    for (int t = 0; t < 4; ++t) {
        if (BK1) {
            pbT[t] = (GCF2)(bBase + (long long)(n0 + tg + 16 * t) * sBc + tp);
        } else {
            pbS[2 * t]     = (GCF)(bBase + (long long)(2 * (bpr + 4 * t))     * sBr + (n0 + bc));
            pbS[2 * t + 1] = (GCF)(bBase + (long long)(2 * (bpr + 4 * t) + 1) * sBr + (n0 + bc));
        }
    }

    v8f acc[2][2] = {};
    int buf = 0;

    for (int k0 = 0; k0 < K; k0 += 32) {
        // ---- phase 1: gather (independent global loads, one wait) ----
        v2f fA[8];
#pragma unroll
        for (int t = 0; t < 8; ++t) {
            fA[t] = paT[t][0];
            paT[t] += 16;   // advance 32 floats
        }
        v2f   fB[4];
        float fB0[4], fB1[4];
#pragma unroll
        for (int t = 0; t < 4; ++t) {
            if (BK1) {
                fB[t] = pbT[t][0];
                pbT[t] += 16;
            } else {
                fB0[t] = pbS[2 * t][0];
                fB1[t] = pbS[2 * t + 1][0];
                pbS[2 * t]     += 32 * sBr;
                pbS[2 * t + 1] += 32 * sBr;
            }
        }

        // ---- phase 2: convert + LDS stores ----
#pragma unroll
        for (int t = 0; t < 8; ++t)
            *(v2bf*)&As[buf][tg + 16 * t][tp] = (v2bf){(__bf16)fA[t][0], (__bf16)fA[t][1]};
#pragma unroll
        for (int t = 0; t < 4; ++t) {
            if (BK1)
                *(v2bf*)&Bst[buf][tg + 16 * t][tp] = (v2bf){(__bf16)fB[t][0], (__bf16)fB[t][1]};
            else
                *(v2bf*)&Bst[buf][bc][2 * (bpr + 4 * t)] = (v2bf){(__bf16)fB0[t], (__bf16)fB1[t]};
        }
        __syncthreads();

        if (k0 + 32 < K) {
            __builtin_prefetch((const void*)paT[0], 0, 1);
            __builtin_prefetch(BK1 ? (const void*)pbT[0] : (const void*)pbS[0], 0, 1);
        }

        // ---- fragments: A lane run = As[row][hi*8 .. +7] and As[row][16+hi*8 .. +7]
        v16bf af[2], bfr[2];
#pragma unroll
        for (int mi = 0; mi < 2; ++mi) {
            const int arow = wm * 32 + mi * 16 + lr;
            v8bf a0 = *(const v8bf*)&As[buf][arow][hi * 8];
            v8bf a1 = *(const v8bf*)&As[buf][arow][16 + hi * 8];
            af[mi] = __builtin_shufflevector(a0, a1, 0,1,2,3,4,5,6,7,8,9,10,11,12,13,14,15);
        }
#pragma unroll
        for (int ni = 0; ni < 2; ++ni) {
            const int bcol = wn * 32 + ni * 16 + lr;
            v8bf b0 = *(const v8bf*)&Bst[buf][bcol][hi * 8];
            v8bf b1 = *(const v8bf*)&Bst[buf][bcol][16 + hi * 8];
            bfr[ni] = __builtin_shufflevector(b0, b1, 0,1,2,3,4,5,6,7,8,9,10,11,12,13,14,15);
        }

#pragma unroll
        for (int mi = 0; mi < 2; ++mi)
#pragma unroll
            for (int ni = 0; ni < 2; ++ni)
                acc[mi][ni] = __builtin_amdgcn_wmma_f32_16x16x32_bf16(
                    false, af[mi], false, bfr[ni], (short)0, acc[mi][ni], false, false);

        buf ^= 1;   // double buffer: single barrier per K-step
    }

    // ---- epilogue: VGPR i -> row i (lanes 0-15) / i+8 (lanes 16-31), col = lane%16
    const long long offC = (long long)bz * sCb + (long long)hz * sCh;
#pragma unroll
    for (int mi = 0; mi < 2; ++mi) {
#pragma unroll
        for (int ni = 0; ni < 2; ++ni) {
            const int col = n0 + wn * 32 + ni * 16 + lr;
            const float bv = bias ? bias[col] : 0.f;
            const int row0 = m0 + wm * 32 + mi * 16 + hi * 8;
            GF  pc = (GF)(C + offC + (long long)row0 * sCr + col);
            GCF pr = resid ? (GCF)(resid + offC + (long long)row0 * sCr + col) : (GCF)0;
#pragma unroll
            for (int i = 0; i < 8; ++i) {
                float v = acc[mi][ni][i] + bv;
                if (act == 1) v = 0.5f * v * (1.f + erff(v * 0.7071067811865475f));
                if (pr) { v += pr[0]; pr += sCr; }
                pc[0] = v;
                pc += sCr;
            }
        }
    }
}

// ---------------- host-side launch helper ----------------
static inline void launch_gemm(hipStream_t stream,
                               const float* A, const float* Bm, const float* bias,
                               const float* resid, float* C,
                               int M, int N, int K, int Hdim, int batchZ, bool bk1,
                               long long sAb, long long sAh, long long sAr,
                               long long sBb, long long sBh, long long sBr, long long sBc,
                               long long sCb, long long sCh, long long sCr,
                               int act)
{
    dim3 grid(N / 64, M / 128, batchZ);
    if (bk1)
        gemm_wmma_bf16<true><<<grid, 256, 0, stream>>>(A, Bm, bias, resid, C, K, Hdim,
                                                       sAb, sAh, sAr,
                                                       sBb, sBh, sBr, sBc,
                                                       sCb, sCh, sCr, act);
    else
        gemm_wmma_bf16<false><<<grid, 256, 0, stream>>>(A, Bm, bias, resid, C, K, Hdim,
                                                        sAb, sAh, sAr,
                                                        sBb, sBh, sBr, sBc,
                                                        sCb, sCh, sCr, act);
}

extern "C" void kernel_launch(void* const* d_in, const int* in_sizes, int n_in,
                              void* d_out, int out_size, void* d_ws, size_t ws_size,
                              hipStream_t stream)
{
    (void)in_sizes; (void)n_in; (void)out_size; (void)ws_size;

    const int*   x      = (const int*)  d_in[0];
    const float* embed  = (const float*)d_in[1];
    const float* Win    = (const float*)d_in[2];
    const float* b_in   = (const float*)d_in[3];
    const float* ln1_g  = (const float*)d_in[4];
    const float* ln1_b  = (const float*)d_in[5];
    const float* Wq     = (const float*)d_in[6];
    const float* bq     = (const float*)d_in[7];
    const float* Wk     = (const float*)d_in[8];
    const float* bk     = (const float*)d_in[9];
    const float* Wv     = (const float*)d_in[10];
    const float* bv     = (const float*)d_in[11];
    const float* Wo     = (const float*)d_in[12];
    const float* bo     = (const float*)d_in[13];
    const float* relE   = (const float*)d_in[14];
    const float* ln2_g  = (const float*)d_in[15];
    const float* ln2_b  = (const float*)d_in[16];
    const float* W1     = (const float*)d_in[17];
    const float* b1     = (const float*)d_in[18];
    const float* W2     = (const float*)d_in[19];
    const float* b2     = (const float*)d_in[20];
    const float* lnf_g  = (const float*)d_in[21];
    const float* lnf_b  = (const float*)d_in[22];
    const float* Wout   = (const float*)d_in[23];
    const float* bout   = (const float*)d_in[24];
    float* out          = (float*)d_out;

    // workspace carve-up (floats)
    const long long BS   = (long long)Bb * Ss;          // 2048
    const long long BSE  = BS * Ee;
    const long long BSH  = BS * HIDh;
    const long long BSF  = BS * Ff;
    const long long QREL = BS * Hh * RP;
    float* ws     = (float*)d_ws;
    float* xe     = ws;               // [B*S, E]
    float* h0     = xe     + BSE;     // residual stream
    float* h1     = h0     + BSH;     // post-attention residual
    float* an     = h1     + BSH;     // LN output
    float* q      = an     + BSH;
    float* k      = q      + BSH;
    float* v      = k      + BSH;
    float* o      = v      + BSH;     // attention output [B,S,H*D]
    float* ff     = o      + BSH;     // [B*S, F]
    float* qrel   = ff     + BSF;     // [B*S*H, RP]
    float* relpad = qrel   + QREL;    // [L, RP, D] zero padded
    float* scs    = relpad + (long long)Ll * RP * Dd;   // [B*H, S, S]

    embed_pe_kernel<<<(unsigned)((BSE + 255) / 256), 256, 0, stream>>>(x, embed, xe);
    relpad_kernel<<<(Ll * RP * Dd + 255) / 256, 256, 0, stream>>>(relE, relpad);

    // input projection: h0 = xe @ Win + b_in   [2048,256]x[256,512]
    launch_gemm(stream, xe, Win, b_in, nullptr, h0,
                (int)BS, HIDh, Ee, 1, 1, false,
                0, 0, Ee,   0, 0, HIDh, 1,   0, 0, HIDh, 0);

    for (int l = 0; l < Ll; ++l) {
        const float* g1 = ln1_g + (long long)l * HIDh;
        const float* c1 = ln1_b + (long long)l * HIDh;
        const float* wq = Wq + (long long)l * HIDh * HIDh;
        const float* wk = Wk + (long long)l * HIDh * HIDh;
        const float* wv = Wv + (long long)l * HIDh * HIDh;
        const float* wo = Wo + (long long)l * HIDh * HIDh;
        const float* rl = relpad + (long long)l * RP * Dd;
        const float* g2 = ln2_g + (long long)l * HIDh;
        const float* c2 = ln2_b + (long long)l * HIDh;
        const float* w1 = W1 + (long long)l * HIDh * Ff;
        const float* w2 = W2 + (long long)l * Ff * HIDh;

        layernorm_kernel<<<(unsigned)BS, 256, 0, stream>>>(h0, g1, c1, an, HIDh);

        // Q, K, V projections [2048,512]x[512,512]
        launch_gemm(stream, an, wq, bq + (long long)l * HIDh, nullptr, q,
                    (int)BS, HIDh, HIDh, 1, 1, false,
                    0, 0, HIDh,  0, 0, HIDh, 1,  0, 0, HIDh, 0);
        launch_gemm(stream, an, wk, bk + (long long)l * HIDh, nullptr, k,
                    (int)BS, HIDh, HIDh, 1, 1, false,
                    0, 0, HIDh,  0, 0, HIDh, 1,  0, 0, HIDh, 0);
        launch_gemm(stream, an, wv, bv + (long long)l * HIDh, nullptr, v,
                    (int)BS, HIDh, HIDh, 1, 1, false,
                    0, 0, HIDh,  0, 0, HIDh, 1,  0, 0, HIDh, 0);

        // qrel[m, r] = q_row(m) . relT[r,:]  (Shaw trick), N padded to 320
        launch_gemm(stream, q, rl, nullptr, nullptr, qrel,
                    (int)(BS * Hh), RP, Dd, 1, 1, true,
                    0, 0, Dd,   0, 0, 1, Dd,   0, 0, RP, 0);

        // scores[z][q,k] = Q[b,:,h,:] @ K[b,:,h,:]^T,  z = b*H + h (B k-contiguous)
        launch_gemm(stream, q, k, nullptr, nullptr, scs,
                    Ss, Ss, Dd, Hh, Bb * Hh, true,
                    (long long)Ss * HIDh, Dd, HIDh,
                    (long long)Ss * HIDh, Dd, 1, HIDh,
                    (long long)Hh * Ss * Ss, (long long)Ss * Ss, Ss, 0);

        softmax_rel_kernel<<<dim3(Ss, Bb * Hh), 256, 0, stream>>>(scs, qrel);

        // o[b, q, h, d] = probs[z] @ V[b,:,h,:]   (B n-contiguous)
        launch_gemm(stream, scs, v, nullptr, nullptr, o,
                    Ss, Dd, Ss, Hh, Bb * Hh, false,
                    (long long)Hh * Ss * Ss, (long long)Ss * Ss, Ss,
                    (long long)Ss * HIDh, Dd, HIDh, 1,
                    (long long)Ss * HIDh, Dd, HIDh, 0);

        // h1 = h0 + o @ Wo + bo
        launch_gemm(stream, o, wo, bo + (long long)l * HIDh, h0, h1,
                    (int)BS, HIDh, HIDh, 1, 1, false,
                    0, 0, HIDh,  0, 0, HIDh, 1,  0, 0, HIDh, 0);

        // LN2 + FFN: h0 = h1 + gelu(ln(h1)@W1 + b1) @ W2 + b2
        layernorm_kernel<<<(unsigned)BS, 256, 0, stream>>>(h1, g2, c2, an, HIDh);
        launch_gemm(stream, an, w1, b1 + (long long)l * Ff, nullptr, ff,
                    (int)BS, Ff, HIDh, 1, 1, false,
                    0, 0, HIDh,  0, 0, Ff, 1,  0, 0, Ff, 1 /*GELU*/);
        launch_gemm(stream, ff, w2, b2 + (long long)l * HIDh, h1, h0,
                    (int)BS, HIDh, Ff, 1, 1, false,
                    0, 0, Ff,  0, 0, HIDh, 1,  0, 0, HIDh, 0);
    }

    // final LN + vocab projection (dominant 67 GFLOP GEMM)
    layernorm_kernel<<<(unsigned)BS, 256, 0, stream>>>(h0, lnf_g, lnf_b, an, HIDh);
    launch_gemm(stream, an, Wout, bout, nullptr, out,
                (int)BS, Vv, HIDh, 1, 1, false,
                0, 0, HIDh,  0, 0, Vv, 1,  0, 0, Vv, 0);
}
